// VariableSelectionNetwork_67723044323584
// MI455X (gfx1250) — matrix operand order repaired and checked by reading, not compiled, not run on hardware
//
#include <hip/hip_runtime.h>

// ---------------------------------------------------------------------------
// MI455X (gfx1250) Variable Selection Network forward.
// Dominant cost: 8192x4096x8192 gate GEMM (~5.5e11 FLOP) -> compute bound.
// bf16 WMMA (v_wmma_f32_16x16x32_bf16, f32 accum) for all GEMMs; bf16 weights
// (~70MB) live in the 192MB L2. All fragments (A and B) are loaded from LDS
// as contiguous 16B runs (ds_load_b128); B is staged transposed
// (fragment-major) with padded stride. s2 demonstrates the TDM path
// (tensor_load_to_lds + s_wait_tensorcnt).
// ---------------------------------------------------------------------------

typedef __attribute__((ext_vector_type(16))) __bf16        v16bf;
typedef __attribute__((ext_vector_type(8)))  float         v8f;
typedef __attribute__((ext_vector_type(4)))  unsigned int  v4u;
typedef __attribute__((ext_vector_type(8)))  int           v8i;
typedef __attribute__((ext_vector_type(4)))  int           v4i;

constexpr int BATCH = 8192;
constexpr int FEAT  = 32;
constexpr int DIM   = 128;
constexpr int HID   = 128;
constexpr int FDIM  = 4096;   // FEAT*DIM
constexpr int FD2   = 8192;   // 2*FDIM

// ---------------- scalar helpers ----------------
__device__ __forceinline__ unsigned short f2bf(float f) {
  unsigned int u = __float_as_uint(f);
  return (unsigned short)((u + 0x7fffu + ((u >> 16) & 1u)) >> 16); // RNE
}
__device__ __forceinline__ float bf2f(unsigned short x) {
  return __uint_as_float(((unsigned int)x) << 16);
}
__device__ __forceinline__ __bf16 bfbits(unsigned short x) {
  union { unsigned short u; __bf16 b; } c; c.u = x; return c.b;
}
__device__ __forceinline__ float eluf(float x) { return x > 0.f ? x : (__expf(x) - 1.f); }
__device__ __forceinline__ float sigmoidf(float x) { return 1.f / (1.f + __expf(-x)); }

// ---------------- WMMA helpers (CDNA5 wave32) ----------------
__device__ __forceinline__ v8f wmma_bf16(v16bf a, v16bf b, v8f c) {
  return __builtin_amdgcn_wmma_f32_16x16x32_bf16(false, a, false, b, (short)0, c, false, false);
}

// A tile: 16(M) x 32(K), row-major (ld elements). ISA 7.12.2: lanes 0-15 hold
// row M=lane, K{0..7,16..23}; lanes 16-31 hold K{8..15,24..31}.
// -> two contiguous 16B runs per lane (2x ds_load_b128).
__device__ __forceinline__ v16bf load_a_frag(const unsigned short* base, int ld, int lane) {
  int row = lane & 15;
  int kh  = (lane >> 4) << 3;
  const unsigned short* p = base + row * ld;
  v16bf a;
#pragma unroll
  for (int i = 0; i < 8; ++i) {
    a[i]     = bfbits(p[kh + i]);
    a[i + 8] = bfbits(p[16 + kh + i]);
  }
  return a;
}

// B tile 32(K) x 16(N) from a TRANSPOSED (fragment-major) LDS image:
// BT[n][k], row stride = STRIDE u16 (16B aligned). lane n = lane&15 needs
// k = kb..kb+7 and 16+kb..16+kb+7 (kb = 8*(lane>>4)) -> two 16B runs (2x ds_load_b128).
__device__ __forceinline__ v16bf load_bT_frag(const unsigned short* baseT, int stride, int lane) {
  const unsigned short* p = baseT + (lane & 15) * stride + ((lane >> 4) << 3);
  v16bf b;
#pragma unroll
  for (int i = 0; i < 8; ++i) {
    b[i]     = bfbits(p[i]);
    b[i + 8] = bfbits(p[16 + i]);
  }
  return b;
}
// C/D layout: lane -> N = lane&15; row = r + 8*(lane>>4), r = vgpr index.

// Stage a KR x 128 row-major bf16 global tile (leading dim ldg) into a
// transposed LDS image BT[n][k] with padded stride. Coalesced u64 reads.
template <int KR, int STRIDE>
__device__ __forceinline__ void stage_bT(const unsigned short* __restrict__ g, int ldg,
                                         unsigned short* bt, int tid) {
  const int total = KR * 32;  // u64 words: KR rows * 128 cols / 4
  for (int u = tid; u < total; u += 256) {
    int k = u >> 5, c = u & 31;
    unsigned long long w = *((const unsigned long long*)(g + (size_t)k * ldg) + c);
    int n = c * 4;
    bt[(n + 0) * STRIDE + k] = (unsigned short)(w);
    bt[(n + 1) * STRIDE + k] = (unsigned short)(w >> 16);
    bt[(n + 2) * STRIDE + k] = (unsigned short)(w >> 32);
    bt[(n + 3) * STRIDE + k] = (unsigned short)(w >> 48);
  }
}

// ---------------- kernel 0: f32 -> bf16 weight conversion ----------------
__global__ void cvt_bf16_kernel(const float* __restrict__ src,
                                unsigned short* __restrict__ dst, int n) {
  int i = blockIdx.x * blockDim.x + threadIdx.x;
  if (i < n) dst[i] = f2bf(src[i]);
}

// ---------------- kernel 1: per-feature GRN + LayerNorm ----------------
// Block: feature f x 64 rows. 256 threads = 8 waves. Dyn LDS = 100KB.
__global__ void fgrn_kernel(const float* __restrict__ x,
                            const float* __restrict__ Wp, const float* __restrict__ bp,
                            const unsigned short* __restrict__ fW1b, const float* __restrict__ fb1,
                            const unsigned short* __restrict__ fW2b, const float* __restrict__ fb2,
                            const unsigned short* __restrict__ fWgb, const float* __restrict__ fbg,
                            const float* __restrict__ fg, const float* __restrict__ fbe,
                            float* __restrict__ flat_f32, unsigned short* __restrict__ flat_b16) {
  extern __shared__ char smem[];
  float*          embf = (float*)smem;                           // 64x128 f32 (32KB)
  unsigned short* actA = (unsigned short*)(smem + 64 * 128 * 4); // 64x128 bf16 (16KB)
  unsigned short* actB = actA + 64 * 128;                        // 64x128 bf16 (16KB)
  unsigned short* WT   = actB + 64 * 128;                        // 128x144 bf16 (36KB)

  const int f    = blockIdx.y;
  const int r0   = blockIdx.x * 64;
  const int tid  = threadIdx.x;
  const int lane = tid & 31;
  const int wave = tid >> 5;
  const int rs   = wave & 3;      // row sub-tile 0..3
  const int half = wave >> 2;     // column half 0..1
  const int n    = lane & 15, m0 = (lane >> 4) * 8;

  // emb = x[:,f,None]*Wp[f] + bp[f]
  for (int i = tid; i < 64 * 128; i += 256) {
    int r = i >> 7, d = i & 127;
    float e = x[(size_t)(r0 + r) * FEAT + f] * Wp[f * DIM + d] + bp[f * DIM + d];
    embf[i] = e;
    actA[i] = f2bf(e);
  }
  stage_bT<128, 144>(fW1b + (size_t)f * DIM * HID, HID, WT, tid);
  __syncthreads();

  // GEMM1: h1 = elu(emb @ fW1[f] + fb1[f]) -> actB
  for (int j = 0; j < 4; ++j) {
    int nt = half * 4 + j;
    v8f acc = {};
    for (int kc = 0; kc < DIM; kc += 32) {
      v16bf a  = load_a_frag(actA + rs * 16 * DIM + kc, DIM, lane);
      v16bf bb = load_bT_frag(WT + nt * 16 * 144 + kc, 144, lane);
      acc = wmma_bf16(a, bb, acc);
    }
    int col = nt * 16 + n;
    float bias = fb1[f * HID + col];
#pragma unroll
    for (int r = 0; r < 8; ++r)
      actB[(rs * 16 + m0 + r) * DIM + col] = f2bf(eluf(acc[r] + bias));
  }
  __syncthreads();
  stage_bT<128, 144>(fW2b + (size_t)f * HID * DIM, DIM, WT, tid);
  __syncthreads();

  // GEMM2: h2 = h1 @ fW2[f] + fb2[f] -> actA
  for (int j = 0; j < 4; ++j) {
    int nt = half * 4 + j;
    v8f acc = {};
    for (int kc = 0; kc < HID; kc += 32) {
      v16bf a  = load_a_frag(actB + rs * 16 * DIM + kc, DIM, lane);
      v16bf bb = load_bT_frag(WT + nt * 16 * 144 + kc, 144, lane);
      acc = wmma_bf16(a, bb, acc);
    }
    int col = nt * 16 + n;
    float bias = fb2[f * DIM + col];
#pragma unroll
    for (int r = 0; r < 8; ++r)
      actA[(rs * 16 + m0 + r) * DIM + col] = f2bf(acc[r] + bias);
  }
  __syncthreads();

  // GEMM3 (paired a|s halves staged through one WT buffer):
  // gates = h2 @ fWg[f] + fbg[f]; emb += a*sigmoid(s)
  v8f accA[4] = {};
  v8f accS[4] = {};
  stage_bT<128, 144>(fWgb + (size_t)f * DIM * 256, 256, WT, tid);  // 'a' columns
  __syncthreads();
  for (int j = 0; j < 4; ++j) {
    int nt = half * 4 + j;
    for (int kc = 0; kc < DIM; kc += 32) {
      v16bf a  = load_a_frag(actA + rs * 16 * DIM + kc, DIM, lane);
      v16bf bb = load_bT_frag(WT + nt * 16 * 144 + kc, 144, lane);
      accA[j] = wmma_bf16(a, bb, accA[j]);
    }
  }
  __syncthreads();
  stage_bT<128, 144>(fWgb + (size_t)f * DIM * 256 + 128, 256, WT, tid);  // 's' columns
  __syncthreads();
  for (int j = 0; j < 4; ++j) {
    int nt = half * 4 + j;
    for (int kc = 0; kc < DIM; kc += 32) {
      v16bf a  = load_a_frag(actA + rs * 16 * DIM + kc, DIM, lane);
      v16bf bb = load_bT_frag(WT + nt * 16 * 144 + kc, 144, lane);
      accS[j] = wmma_bf16(a, bb, accS[j]);
    }
  }
  for (int j = 0; j < 4; ++j) {
    int nt = half * 4 + j;
    int col = nt * 16 + n;
    float bA = fbg[f * 256 + col];
    float bS = fbg[f * 256 + 128 + col];
#pragma unroll
    for (int r = 0; r < 8; ++r) {
      int m = rs * 16 + m0 + r;
      embf[m * DIM + col] += (accA[j][r] + bA) * sigmoidf(accS[j][r] + bS);
    }
  }
  __syncthreads();

  // LayerNorm over D=128, write flat (f32 + bf16). 8 rows per wave.
  for (int rr = 0; rr < 8; ++rr) {
    int r = wave * 8 + rr;
    float s = 0.f, s2 = 0.f;
    for (int d = lane; d < 128; d += 32) { float v = embf[r * 128 + d]; s += v; s2 += v * v; }
    for (int off = 16; off; off >>= 1) { s += __shfl_xor(s, off, 32); s2 += __shfl_xor(s2, off, 32); }
    float mu = s * (1.f / 128.f);
    float var = s2 * (1.f / 128.f) - mu * mu;
    float rstd = rsqrtf(var + 1e-5f);
    for (int d = lane; d < 128; d += 32) {
      float v = (embf[r * 128 + d] - mu) * rstd * fg[f * 128 + d] + fbe[f * 128 + d];
      size_t idx = (size_t)(r0 + r) * FDIM + f * 128 + d;
      flat_f32[idx] = v;
      flat_b16[idx] = f2bf(v);
    }
  }
}

// ---------------- kernel 2: hh1 = elu(flat @ sW1 + sb1) ----------------
__global__ void s1_kernel(const unsigned short* __restrict__ flat,
                          const unsigned short* __restrict__ sW1b,
                          const float* __restrict__ sb1,
                          unsigned short* __restrict__ hh1) {
  extern __shared__ char smem[];
  unsigned short* Ach = (unsigned short*)smem;   // 64x32 row-major (4KB)
  unsigned short* BT  = Ach + 64 * 32;           // 128n x 48 stride (12KB)
  const int r0 = blockIdx.x * 64;
  const int tid = threadIdx.x, lane = tid & 31, wave = tid >> 5;
  const int rs = wave & 3, half = wave >> 2;
  v8f acc[4] = {};

  for (int kc = 0; kc < FDIM; kc += 32) {
    for (int u = tid; u < 512; u += 256) {               // A: 64 rows x 8 u64
      int r = u >> 3, c = u & 7;
      ((unsigned long long*)Ach)[u] =
          *((const unsigned long long*)(flat + (size_t)(r0 + r) * FDIM + kc) + c);
    }
    stage_bT<32, 48>(sW1b + (size_t)kc * HID, HID, BT, tid);
    __syncthreads();
    v16bf a = load_a_frag(Ach + rs * 16 * 32, 32, lane);
#pragma unroll
    for (int j = 0; j < 4; ++j) {
      v16bf bb = load_bT_frag(BT + (half * 4 + j) * 16 * 48, 48, lane);
      acc[j] = wmma_bf16(a, bb, acc[j]);
    }
    __syncthreads();
  }
  int n = lane & 15, m0 = (lane >> 4) * 8;
  for (int j = 0; j < 4; ++j) {
    int col = (half * 4 + j) * 16 + n;
    float bias = sb1[col];
#pragma unroll
    for (int r = 0; r < 8; ++r) {
      int m = r0 + rs * 16 + m0 + r;
      hh1[(size_t)m * HID + col] = f2bf(eluf(acc[j][r] + bias));
    }
  }
}

// ---------------- kernel 3: hh2 = hh1 @ sW2 + sb2 ----------------
// A tile loaded by the Tensor Data Mover (gfx1250 TDM) when available.
__global__ void s2_kernel(const unsigned short* __restrict__ hh1,
                          const unsigned short* __restrict__ sW2b,
                          const float* __restrict__ sb2,
                          unsigned short* __restrict__ hh2) {
  extern __shared__ char smem[];
  unsigned short* At = (unsigned short*)smem;   // 64x128 row-major (16KB)
  unsigned short* BT = At + 64 * 128;           // 128n x 144 stride (36KB)
  const int r0 = blockIdx.x * 64, n0 = blockIdx.y * 128;
  const int tid = threadIdx.x, lane = tid & 31, wave = tid >> 5;
  const int rs = wave & 3, half = wave >> 2;

#if defined(__has_builtin) && __has_builtin(__builtin_amdgcn_tensor_load_to_lds)
  if (wave == 0) {
    // TDM 2D tile: 64 rows x 128 cols of hh1 (2-byte elements) -> LDS @ At.
    unsigned lds_addr = (unsigned)(unsigned long long)(void*)At;
    unsigned long long ga = (unsigned long long)(const void*)(hh1 + (size_t)r0 * HID);
    v4u g0;
    g0[0] = 1u;                                          // count=1
    g0[1] = lds_addr;                                    // lds_addr [63:32]
    g0[2] = (unsigned)(ga & 0xffffffffull);              // global_addr low
    g0[3] = (unsigned)((ga >> 32) & 0x1ffffffull) | (2u << 30);  // ga hi | type=2
    v8i g1;
    g1[0] = (int)(1u << 16);                             // data_size = 2B
    g1[1] = (int)((unsigned)HID << 16);                  // tensor_dim0 = 128
    g1[2] = (int)((unsigned)BATCH << 16);                // tensor_dim1 = 8192
    g1[3] = (int)((unsigned)HID << 16);                  // tile_dim0 = 128
    g1[4] = 64;                                          // tile_dim1 = 64
    g1[5] = HID;                                         // tensor_dim0_stride = 128
    g1[6] = 0;
    g1[7] = 0;
    v4i gz = {0, 0, 0, 0};
#if __has_include(<hip/amd_detail/amd_gfx1250_TDM.h>)
    v8i gz8 = {0, 0, 0, 0, 0, 0, 0, 0};
    __builtin_amdgcn_tensor_load_to_lds(g0, g1, gz, gz, gz8, 0);
#else
    __builtin_amdgcn_tensor_load_to_lds(g0, g1, gz, gz, 0);
#endif
    __builtin_amdgcn_s_wait_tensorcnt(0);
  }
#else
  for (int u = tid; u < 2048; u += 256) {       // A: 64 rows x 32 u64
    int r = u >> 5, c = u & 31;
    ((unsigned long long*)At)[u] =
        *((const unsigned long long*)(hh1 + (size_t)(r0 + r) * HID) + c);
  }
#endif
  stage_bT<128, 144>(sW2b + n0, FDIM, BT, tid);
  __syncthreads();

  for (int j = 0; j < 4; ++j) {
    int nt = half * 4 + j;
    v8f acc = {};
    for (int kc = 0; kc < HID; kc += 32) {
      v16bf a  = load_a_frag(At + rs * 16 * HID + kc, HID, lane);
      v16bf bb = load_bT_frag(BT + nt * 16 * 144 + kc, 144, lane);
      acc = wmma_bf16(a, bb, acc);
    }
    int n = lane & 15, m0 = (lane >> 4) * 8;
    int gc = n0 + nt * 16 + n;
    float bias = sb2[gc];
#pragma unroll
    for (int r = 0; r < 8; ++r) {
      int m = r0 + rs * 16 + m0 + r;
      hh2[(size_t)m * FDIM + gc] = f2bf(acc[r] + bias);
    }
  }
}

// ---------------- kernel 4: dominant paired gate GEMM ----------------
// [a|s] = hh2 @ sWg + sbg ; selraw = flat + a*sigmoid(s). Block owns a 64-row
// x 128-col slab of 'a' and the matching slab of 's' (col+FDIM): the gate
// fuses in-register; the 2FD-wide gate tensor never touches memory.
__global__ void s3_kernel(const unsigned short* __restrict__ hh2,
                          const unsigned short* __restrict__ sWgb,
                          const float* __restrict__ sbg,
                          const float* __restrict__ flat_f32,
                          float* __restrict__ selraw) {
  extern __shared__ char smem[];
  unsigned short* Ach = (unsigned short*)smem;  // 64x32 row-major (4KB)
  unsigned short* BTa = Ach + 64 * 32;          // 128n x 48 ('a' cols, 12KB)
  unsigned short* BTs = BTa + 128 * 48;         // 128n x 48 ('s' cols, 12KB)
  const int r0 = blockIdx.x * 64;
  const int j0 = blockIdx.y * 128;
  const int tid = threadIdx.x, lane = tid & 31, wave = tid >> 5;
  const int rs = wave & 3, half = wave >> 2;
  v8f accA[4] = {};
  v8f accS[4] = {};

  for (int kc = 0; kc < FDIM; kc += 32) {
    for (int u = tid; u < 512; u += 256) {      // A chunk: 64 rows x 8 u64
      int r = u >> 3, c = u & 7;
      ((unsigned long long*)Ach)[u] =
          *((const unsigned long long*)(hh2 + (size_t)(r0 + r) * FDIM + kc) + c);
    }
    stage_bT<32, 48>(sWgb + (size_t)kc * FD2 + j0, FD2, BTa, tid);
    stage_bT<32, 48>(sWgb + (size_t)kc * FD2 + j0 + FDIM, FD2, BTs, tid);
    if (kc + 32 < FDIM) {                       // prefetch next weight chunk
      __builtin_prefetch(sWgb + (size_t)(kc + 32 + (tid >> 3)) * FD2 + j0 + (tid & 7) * 16, 0, 0);
    }
    __syncthreads();
    v16bf a = load_a_frag(Ach + rs * 16 * 32, 32, lane);
#pragma unroll
    for (int j = 0; j < 4; ++j) {
      int nt = half * 4 + j;
      v16bf ba = load_bT_frag(BTa + nt * 16 * 48, 48, lane);
      accA[j] = wmma_bf16(a, ba, accA[j]);
      v16bf bs = load_bT_frag(BTs + nt * 16 * 48, 48, lane);
      accS[j] = wmma_bf16(a, bs, accS[j]);
    }
    __syncthreads();
  }

  int n = lane & 15, m0 = (lane >> 4) * 8;
  for (int j = 0; j < 4; ++j) {
    int gcol = j0 + (half * 4 + j) * 16 + n;
    float bA = sbg[gcol];
    float bS = sbg[FDIM + gcol];
#pragma unroll
    for (int r = 0; r < 8; ++r) {
      int m = r0 + rs * 16 + m0 + r;
      size_t idx = (size_t)m * FDIM + gcol;
      selraw[idx] = flat_f32[idx] + (accA[j][r] + bA) * sigmoidf(accS[j][r] + bS);
    }
  }
}

// ---------------- kernel 5: LayerNorm over FD=4096 ----------------
__global__ void ln2_kernel(const float* __restrict__ selraw,
                           const float* __restrict__ sg, const float* __restrict__ sbe,
                           unsigned short* __restrict__ selin_b16) {
  __shared__ float red0[8];
  __shared__ float red1[8];
  const int b = blockIdx.x;
  const int tid = threadIdx.x, lane = tid & 31, wave = tid >> 5;
  const float* row = selraw + (size_t)b * FDIM;
  float s = 0.f, s2 = 0.f;
  for (int i = tid; i < FDIM; i += 256) { float v = row[i]; s += v; s2 += v * v; }
  for (int off = 16; off; off >>= 1) { s += __shfl_xor(s, off, 32); s2 += __shfl_xor(s2, off, 32); }
  if (lane == 0) { red0[wave] = s; red1[wave] = s2; }
  __syncthreads();
  float ts = 0.f, ts2 = 0.f;
#pragma unroll
  for (int i = 0; i < 8; ++i) { ts += red0[i]; ts2 += red1[i]; }
  float mu = ts * (1.f / FDIM);
  float var = ts2 * (1.f / FDIM) - mu * mu;
  float rstd = rsqrtf(var + 1e-5f);
  for (int i = tid; i < FDIM; i += 256) {
    float v = (row[i] - mu) * rstd * sg[i] + sbe[i];
    selin_b16[(size_t)b * FDIM + i] = f2bf(v);
  }
}

// ---------------- kernel 6: head (logits, softmax, weighted sum) ----------------
__global__ void head_kernel(const unsigned short* __restrict__ selin,
                            const float* __restrict__ flat_f32,
                            const float* __restrict__ Wh, const float* __restrict__ bh,
                            float* __restrict__ out_sel, float* __restrict__ out_w) {
  __shared__ float smem[FEAT * 128];
  __shared__ float wsm[FEAT];
  const int b = blockIdx.x;
  const int tid = threadIdx.x;  // 0..127
  float acc[FEAT] = {};
  for (int k = tid; k < FDIM; k += 128) {
    float v = bf2f(selin[(size_t)b * FDIM + k]);
    const float* wr = Wh + (size_t)k * FEAT;
#pragma unroll
    for (int f = 0; f < FEAT; ++f) acc[f] += v * wr[f];
  }
#pragma unroll
  for (int f = 0; f < FEAT; ++f) smem[f * 128 + tid] = acc[f];
  __syncthreads();

  if (tid < 32) {
    float s = 0.f;
#pragma unroll
    for (int i = 0; i < 128; ++i) s += smem[tid * 128 + i];
    float l = s + bh[tid];
    float m = l;
    for (int off = 16; off; off >>= 1) m = fmaxf(m, __shfl_xor(m, off, 32));
    float e = __expf(l - m);
    float se = e;
    for (int off = 16; off; off >>= 1) se += __shfl_xor(se, off, 32);
    float w = e / se;
    out_w[(size_t)b * FEAT + tid] = w;
    wsm[tid] = w;
  }
  __syncthreads();

  float a = 0.f;
#pragma unroll
  for (int f = 0; f < FEAT; ++f) a += wsm[f] * flat_f32[(size_t)b * FDIM + f * 128 + tid];
  out_sel[(size_t)b * DIM + tid] = a;
}

// ---------------- launcher ----------------
extern "C" void kernel_launch(void* const* d_in, const int* in_sizes, int n_in,
                              void* d_out, int out_size, void* d_ws, size_t ws_size,
                              hipStream_t stream) {
  const float* x   = (const float*)d_in[0];
  const float* Wp  = (const float*)d_in[1];
  const float* bp  = (const float*)d_in[2];
  const float* fW1 = (const float*)d_in[3];
  const float* fb1 = (const float*)d_in[4];
  const float* fW2 = (const float*)d_in[5];
  const float* fb2 = (const float*)d_in[6];
  const float* fWg = (const float*)d_in[7];
  const float* fbg = (const float*)d_in[8];
  const float* fg  = (const float*)d_in[9];
  const float* fbe = (const float*)d_in[10];
  const float* sW1 = (const float*)d_in[11];
  const float* sb1 = (const float*)d_in[12];
  const float* sW2 = (const float*)d_in[13];
  const float* sb2 = (const float*)d_in[14];
  const float* sWg = (const float*)d_in[15];
  const float* sbg = (const float*)d_in[16];
  const float* sg  = (const float*)d_in[17];
  const float* sbe = (const float*)d_in[18];
  const float* Wh  = (const float*)d_in[19];
  const float* bh  = (const float*)d_in[20];

  // Workspace layout
  char* p = (char*)d_ws;
  float* flat_f32 = (float*)p;            p += (size_t)BATCH * FDIM * 4;
  float* selraw   = (float*)p;            p += (size_t)BATCH * FDIM * 4;
  unsigned short* flat_b16 = (unsigned short*)p; p += (size_t)BATCH * FDIM * 2; // reused as selin
  unsigned short* hh2_b16  = (unsigned short*)p; p += (size_t)BATCH * FDIM * 2;
  unsigned short* hh1_b16  = (unsigned short*)p; p += (size_t)BATCH * HID * 2;
  unsigned short* fW1b = (unsigned short*)p; p += (size_t)FEAT * DIM * HID * 2;
  unsigned short* fW2b = (unsigned short*)p; p += (size_t)FEAT * HID * DIM * 2;
  unsigned short* fWgb = (unsigned short*)p; p += (size_t)FEAT * DIM * 256 * 2;
  unsigned short* sW1b = (unsigned short*)p; p += (size_t)FDIM * HID * 2;
  unsigned short* sW2b = (unsigned short*)p; p += (size_t)HID * FDIM * 2;
  unsigned short* sWgb = (unsigned short*)p; p += (size_t)FDIM * FD2 * 2;

  float* out_sel = (float*)d_out;
  float* out_w   = (float*)d_out + (size_t)BATCH * DIM;

  auto cvt = [&](const float* src, unsigned short* dst, size_t n) {
    cvt_bf16_kernel<<<dim3((unsigned)((n + 255) / 256)), dim3(256), 0, stream>>>(src, dst, (int)n);
  };
  cvt(fW1, fW1b, (size_t)FEAT * DIM * HID);
  cvt(fW2, fW2b, (size_t)FEAT * HID * DIM);
  cvt(fWg, fWgb, (size_t)FEAT * DIM * 256);
  cvt(sW1, sW1b, (size_t)FDIM * HID);
  cvt(sW2, sW2b, (size_t)HID * FDIM);
  cvt(sWg, sWgb, (size_t)FDIM * FD2);

  fgrn_kernel<<<dim3(BATCH / 64, FEAT), dim3(256),
                (64 * 128 * 4) + (64 * 128 * 2) * 2 + (128 * 144 * 2), stream>>>(
      x, Wp, bp, fW1b, fb1, fW2b, fb2, fWgb, fbg, fg, fbe, flat_f32, flat_b16);

  s1_kernel<<<dim3(BATCH / 64), dim3(256), (64 * 32 + 128 * 48) * 2, stream>>>(
      flat_b16, sW1b, sb1, hh1_b16);

  s2_kernel<<<dim3(BATCH / 64, FDIM / 128), dim3(256), (64 * 128 + 128 * 144) * 2, stream>>>(
      hh1_b16, sW2b, sb2, hh2_b16);

  s3_kernel<<<dim3(BATCH / 64, FDIM / 128), dim3(256), (64 * 32 + 2 * 128 * 48) * 2, stream>>>(
      hh2_b16, sWgb, sbg, flat_f32, selraw);

  ln2_kernel<<<dim3(BATCH), dim3(256), 0, stream>>>(selraw, sg, sbe, flat_b16);

  head_kernel<<<dim3(BATCH), dim3(128), 0, stream>>>(flat_b16, flat_f32, Wh, bh, out_sel, out_w);
}
